// EncoderBlock_68006512165391
// MI455X (gfx1250) — compile-verified, matrix operand order checked by reading, not run on hardware
//
#include <hip/hip_runtime.h>

// ---------------------------------------------------------------------------
// Problem constants (match reference)
// ---------------------------------------------------------------------------
#define BB 8
#define SS 1024
#define DM 1024
#define NH 16
#define DKH 64
#define DFF 4096
#define RR (BB * SS)          // 8192 rows
#define C3 (3 * DM)           // 3072 (QKV concat)

typedef __attribute__((ext_vector_type(16))) _Float16 v16h;
typedef __attribute__((ext_vector_type(8)))  _Float16 v8h;
typedef __attribute__((ext_vector_type(4)))  _Float16 v4h;
typedef __attribute__((ext_vector_type(8)))  float    v8f;

union Frag16 {
    v16h v;
    v8h  h2[2];
    _Float16 e[16];
};

static __device__ __forceinline__ v8f wmma_f16(v16h a, v16h b, v8f c) {
    return __builtin_amdgcn_wmma_f32_16x16x32_f16(
        /*neg_a=*/false, a, /*neg_b=*/false, b,
        /*c_mod=*/(short)0, c, /*reuse_a=*/false, /*reuse_b=*/false);
}

// ---------------------------------------------------------------------------
// CDNA5 async global->LDS copy (ASYNCcnt tracked). Inline asm per bridge doc:
// portable across ROCm-7.2 / amdgpu-toolchain (builtin arity differs).
// ---------------------------------------------------------------------------
#define USE_ASYNC_COPY 1

typedef __attribute__((address_space(3))) _Float16 lds_f16;

static __device__ __forceinline__ void cp_b128(const _Float16* g, _Float16* l) {
#if USE_ASYNC_COPY
    unsigned loff = (unsigned)(size_t)(lds_f16*)l;     // 0-based LDS byte addr
    unsigned long long ga = (unsigned long long)(size_t)g;
    asm volatile("global_load_async_to_lds_b128 %0, %1, off"
                 :: "v"(loff), "v"(ga) : "memory");
#else
    *(uint4*)l = *(const uint4*)g;
#endif
}

static __device__ __forceinline__ void wait_async() {
#if USE_ASYNC_COPY
    asm volatile("s_wait_asynccnt 0x0" ::: "memory");
#endif
}

// ---------------------------------------------------------------------------
// Weight packing: f32 [K][N] -> f16 transposed [N][K] (one-time, reused by
// all 64 M-blocks, so GEMM staging becomes pure 128-bit copies)
// ---------------------------------------------------------------------------
__global__ void cvt_t_kernel(const float* __restrict__ src,   // [K][N]
                             _Float16* __restrict__ dst,      // [N][K]
                             int K, int N) {
    int i = blockIdx.x * 256 + threadIdx.x;                   // over N*K
    int n = i / K;
    int k = i - n * K;
    dst[i] = (_Float16)src[(size_t)k * N + n];
}

__global__ void pack_qkv_w_kernel(const float* __restrict__ wq,
                                  const float* __restrict__ wk,
                                  const float* __restrict__ wv,
                                  _Float16* __restrict__ dst) {  // [3072][1024]
    int i = blockIdx.x * 256 + threadIdx.x;                      // over 3072*1024
    int n = i >> 10;
    int k = i & (DM - 1);
    float v;
    if (n < DM)            v = wq[(size_t)k * DM + n];
    else if (n < 2 * DM)   v = wk[(size_t)k * DM + (n - DM)];
    else                   v = wv[(size_t)k * DM + (n - 2 * DM)];
    dst[i] = (_Float16)v;
}

__global__ void pack_qkv_b_kernel(const float* __restrict__ bq,
                                  const float* __restrict__ bk,
                                  const float* __restrict__ bv,
                                  float* __restrict__ dst) {
    int n = blockIdx.x * 256 + threadIdx.x;      // over 3072
    if (n < DM)            dst[n] = bq[n];
    else if (n < 2 * DM)   dst[n] = bk[n - DM];
    else if (n < C3)       dst[n] = bv[n - 2 * DM];
}

// ---------------------------------------------------------------------------
// LayerNorm (ddof=1, eps added to std) -> f16 output for WMMA consumption
// ---------------------------------------------------------------------------
__global__ __launch_bounds__(256)
void ln_kernel(const float* __restrict__ x,
               const float* __restrict__ alpha,
               const float* __restrict__ beta,
               _Float16* __restrict__ out) {
    __shared__ float red[256];
    __shared__ float red2[256];
    const int row = blockIdx.x;
    const int tid = threadIdx.x;
    const float4 v = *(const float4*)(x + (size_t)row * DM + tid * 4);
    float s  = v.x + v.y + v.z + v.w;
    float s2 = v.x * v.x + v.y * v.y + v.z * v.z + v.w * v.w;
    red[tid] = s; red2[tid] = s2;
    __syncthreads();
    for (int off = 128; off > 0; off >>= 1) {
        if (tid < off) { red[tid] += red[tid + off]; red2[tid] += red2[tid + off]; }
        __syncthreads();
    }
    const float mean = red[0] * (1.0f / DM);
    const float var  = (red2[0] - (float)DM * mean * mean) * (1.0f / (DM - 1));
    const float inv  = 1.0f / (sqrtf(var) + 1e-6f);
    const float a = alpha[0], bb = beta[0];
    v4h o;
    o[0] = (_Float16)(a * (v.x - mean) * inv + bb);
    o[1] = (_Float16)(a * (v.y - mean) * inv + bb);
    o[2] = (_Float16)(a * (v.z - mean) * inv + bb);
    o[3] = (_Float16)(a * (v.w - mean) * inv + bb);
    *(v4h*)(out + (size_t)row * DM + tid * 4) = o;
}

// ---------------------------------------------------------------------------
// WMMA GEMM: C[M,N] = A[M,K](f16) * Bt[N,K](f16) + bias. Double-buffered LDS,
// async-to-LDS staging. BM=BN=128, BK=32, 8 waves, each wave 32x64.
// QKVMODE: N=3072; cols [0,2048) -> qk buffer (stride 2048), cols [2048,3072)
// -> transposed V buffer vt[b][h][d][s] (free transpose in epilogue).
// ---------------------------------------------------------------------------
template <bool RELU, bool RESID, bool OUTF32, bool QKVMODE>
__global__ __launch_bounds__(256)
void gemm_wmma_kernel(const _Float16* __restrict__ A,
                      const _Float16* __restrict__ Bt,
                      const float* __restrict__ bias,
                      const float* __restrict__ resid,
                      void* __restrict__ Cout,
                      _Float16* __restrict__ vt,
                      int M, int N, int K) {
    constexpr int BM = 128, BN = 128, BK = 32, LDT = 40;  // 80B rows (16B aligned)
    __shared__ __align__(16) _Float16 sA[2][BM * LDT];
    __shared__ __align__(16) _Float16 sB[2][BN * LDT];

    const int tid  = threadIdx.x;
    const int lane = tid & 31;
    const int wid  = tid >> 5;
    const int bm0  = blockIdx.y * BM;
    const int bn0  = blockIdx.x * BN;
    const int wm   = (wid & 3) * 32;
    const int wn   = (wid >> 2) * 64;
    const int lrow = lane & 15;
    const int kb8  = (lane < 16) ? 0 : 8;

    v8f acc[2][4];
#pragma unroll
    for (int mi = 0; mi < 2; ++mi)
#pragma unroll
        for (int ni = 0; ni < 4; ++ni)
#pragma unroll
            for (int i = 0; i < 8; ++i) acc[mi][ni][i] = 0.0f;

    const int r0 = tid >> 2;          // 0..63
    const int c0 = (tid & 3) * 8;     // 0,8,16,24

    auto stage = [&](int buf, int k0) {
#pragma unroll
        for (int it = 0; it < 2; ++it) {
            const int r = r0 + it * 64;
            cp_b128(A  + (size_t)(bm0 + r) * K + k0 + c0, &sA[buf][r * LDT + c0]);
            cp_b128(Bt + (size_t)(bn0 + r) * K + k0 + c0, &sB[buf][r * LDT + c0]);
        }
    };

    stage(0, 0);
    wait_async();
    __syncthreads();

    int buf = 0;
    for (int k0 = 0; k0 < K; k0 += BK) {
        const bool more = (k0 + BK) < K;
        if (more) stage(buf ^ 1, k0 + BK);   // overlaps with WMMA below

        Frag16 af[2], bf[4];
#pragma unroll
        for (int mi = 0; mi < 2; ++mi) {
            const int r = wm + mi * 16 + lrow;
            af[mi].h2[0] = *(const v8h*)(&sA[buf][r * LDT + kb8]);
            af[mi].h2[1] = *(const v8h*)(&sA[buf][r * LDT + kb8 + 16]);
        }
#pragma unroll
        for (int ni = 0; ni < 4; ++ni) {
            const int c = wn + ni * 16 + lrow;
            bf[ni].h2[0] = *(const v8h*)(&sB[buf][c * LDT + kb8]);
            bf[ni].h2[1] = *(const v8h*)(&sB[buf][c * LDT + kb8 + 16]);
        }
#pragma unroll
        for (int mi = 0; mi < 2; ++mi)
#pragma unroll
            for (int ni = 0; ni < 4; ++ni)
                acc[mi][ni] = wmma_f16(af[mi].v, bf[ni].v, acc[mi][ni]);

        if (more) wait_async();
        __syncthreads();
        buf ^= 1;
    }

    // Epilogue: C layout: lane -> col (lane&15); VGPR r -> row r + 8*(lane>=16).
    const int rbase = (lane < 16) ? 0 : 8;
#pragma unroll
    for (int mi = 0; mi < 2; ++mi)
#pragma unroll
        for (int ni = 0; ni < 4; ++ni) {
            const int col = bn0 + wn + ni * 16 + lrow;
            const float bv = bias ? bias[col] : 0.0f;
#pragma unroll
            for (int r = 0; r < 8; ++r) {
                const int row = bm0 + wm + mi * 16 + rbase + r;
                float v = acc[mi][ni][r] + bv;
                if (RELU) v = fmaxf(v, 0.0f);
                if (RESID) v += resid[(size_t)row * N + col];
                if (QKVMODE) {
                    if (col < 2 * DM) {
                        ((_Float16*)Cout)[(size_t)row * (2 * DM) + col] = (_Float16)v;
                    } else {
                        const int d = col - 2 * DM;          // h*64 + dv
                        const int bI = row >> 10;
                        const int s  = row & (SS - 1);
                        vt[((size_t)bI * DM + d) * SS + s] = (_Float16)v;
                    }
                } else if (OUTF32) {
                    ((float*)Cout)[(size_t)row * N + col] = v;
                } else {
                    ((_Float16*)Cout)[(size_t)row * N + col] = (_Float16)v;
                }
            }
        }
}

// ---------------------------------------------------------------------------
// Flash-attention: block = 128 threads = 4 waves; each wave owns one 16-row
// Q tile of one (b,h). qk: f16 [R, 2048] (q|k), vt: f16 [B*H*64][S].
// ---------------------------------------------------------------------------
__global__ __launch_bounds__(128)
void attn_kernel(const _Float16* __restrict__ qk,
                 const _Float16* __restrict__ vt,
                 const int* __restrict__ mask,
                 _Float16* __restrict__ o) {
    __shared__ __align__(16) _Float16 pbuf[4][16 * 32];  // per-wave P tile (16x32)

    const int tid  = threadIdx.x;
    const int lane = tid & 31;
    const int w    = tid >> 5;
    const int bid  = blockIdx.x;             // B*H*(S/64) = 2048
    const int qt   = bid & (SS / 64 - 1);
    const int h    = (bid >> 4) & (NH - 1);
    const int b    = bid >> 8;
    const int q0   = qt * 64 + w * 16;
    const int lrow  = lane & 15;
    const int kb8   = (lane < 16) ? 0 : 8;
    const int rbase = (lane < 16) ? 0 : 8;

    // Q as two A-fragments (head dim 64 = 2 x K32), loaded once.
    Frag16 qa[2];
    {
        const size_t roff = (size_t)(b * SS + q0 + lrow) * (2 * DM) + h * DKH;
#pragma unroll
        for (int f = 0; f < 2; ++f) {
            qa[f].h2[0] = *(const v8h*)(qk + roff + f * 32 + kb8);
            qa[f].h2[1] = *(const v8h*)(qk + roff + f * 32 + kb8 + 16);
        }
    }

    float mrow[8], ssum[8];
    v8f oacc[4];
#pragma unroll
    for (int r = 0; r < 8; ++r) { mrow[r] = -3.0e38f; ssum[r] = 0.0f; }
#pragma unroll
    for (int c = 0; c < 4; ++c)
#pragma unroll
        for (int i = 0; i < 8; ++i) oacc[c][i] = 0.0f;

    for (int kb = 0; kb < SS; kb += 32) {
        // ---- scores for two 16-key tiles ----
        v8f sc[2];
#pragma unroll
        for (int t = 0; t < 2; ++t) {
            const int key = kb + t * 16 + lrow;
            const size_t koff = (size_t)(b * SS + key) * (2 * DM) + DM + h * DKH;
            Frag16 kf0, kf1;
            kf0.h2[0] = *(const v8h*)(qk + koff + kb8);
            kf0.h2[1] = *(const v8h*)(qk + koff + kb8 + 16);
            kf1.h2[0] = *(const v8h*)(qk + koff + 32 + kb8);
            kf1.h2[1] = *(const v8h*)(qk + koff + 32 + kb8 + 16);
            v8f z;
#pragma unroll
            for (int i = 0; i < 8; ++i) z[i] = 0.0f;
            z     = wmma_f16(qa[0].v, kf0.v, z);
            sc[t] = wmma_f16(qa[1].v, kf1.v, z);
            const int mk = mask[b * SS + key];
#pragma unroll
            for (int r = 0; r < 8; ++r) {
                const float v = sc[t][r] * 0.125f;     // 1/sqrt(64)
                sc[t][r] = (mk == 0) ? -1.0e9f : v;
            }
        }
        // ---- online softmax per row (row spans 16 lanes of a half-wave) ----
#pragma unroll
        for (int r = 0; r < 8; ++r) {
            float tmax = fmaxf(sc[0][r], sc[1][r]);
            for (int off = 8; off >= 1; off >>= 1)
                tmax = fmaxf(tmax, __shfl_xor(tmax, off, 32));
            const float newm = fmaxf(mrow[r], tmax);
            const float corr = __expf(mrow[r] - newm);
            const float p0 = __expf(sc[0][r] - newm);
            const float p1 = __expf(sc[1][r] - newm);
            float ps = p0 + p1;
            for (int off = 8; off >= 1; off >>= 1)
                ps += __shfl_xor(ps, off, 32);
            ssum[r] = ssum[r] * corr + ps;
            mrow[r] = newm;
#pragma unroll
            for (int c = 0; c < 4; ++c) oacc[c][r] *= corr;
            pbuf[w][(rbase + r) * 32 + lrow]      = (_Float16)p0;
            pbuf[w][(rbase + r) * 32 + 16 + lrow] = (_Float16)p1;
        }
        __syncthreads();
        Frag16 pa;
        pa.h2[0] = *(const v8h*)(&pbuf[w][lrow * 32 + kb8]);
        pa.h2[1] = *(const v8h*)(&pbuf[w][lrow * 32 + kb8 + 16]);
        __syncthreads();
        // ---- accumulate P(16x32) @ V(32x64): V from transposed vt ----
#pragma unroll
        for (int c = 0; c < 4; ++c) {
            const size_t vrow = ((size_t)b * DM + h * DKH + c * 16 + lrow) * SS + kb;
            Frag16 vf;
            vf.h2[0] = *(const v8h*)(vt + vrow + kb8);
            vf.h2[1] = *(const v8h*)(vt + vrow + kb8 + 16);
            oacc[c] = wmma_f16(pa.v, vf.v, oacc[c]);
        }
    }
    // ---- normalize and store f16 output ----
#pragma unroll
    for (int c = 0; c < 4; ++c)
#pragma unroll
        for (int r = 0; r < 8; ++r) {
            const int row = b * SS + q0 + rbase + r;
            o[(size_t)row * DM + h * DKH + c * 16 + lrow] =
                (_Float16)(oacc[c][r] / ssum[r]);
        }
}

// ---------------------------------------------------------------------------
// Host-side orchestration (all on `stream`; graph-capture safe)
// ---------------------------------------------------------------------------
extern "C" void kernel_launch(void* const* d_in, const int* in_sizes, int n_in,
                              void* d_out, int out_size, void* d_ws, size_t ws_size,
                              hipStream_t stream) {
    const float* x    = (const float*)d_in[0];
    const int*   mask = (const int*)d_in[1];
    const float* wq   = (const float*)d_in[2];
    const float* bq   = (const float*)d_in[3];
    const float* wk   = (const float*)d_in[4];
    const float* bk   = (const float*)d_in[5];
    const float* wv   = (const float*)d_in[6];
    const float* bv   = (const float*)d_in[7];
    const float* wo   = (const float*)d_in[8];
    const float* bo   = (const float*)d_in[9];
    const float* w1   = (const float*)d_in[10];
    const float* b1   = (const float*)d_in[11];
    const float* w2   = (const float*)d_in[12];
    const float* b2   = (const float*)d_in[13];
    const float* ln1a = (const float*)d_in[14];
    const float* ln1b = (const float*)d_in[15];
    const float* ln2a = (const float*)d_in[16];
    const float* ln2b = (const float*)d_in[17];

    // Workspace layout (bytes, 16B aligned chunks)
    char* ws = (char*)d_ws;
    size_t off = 0;
    _Float16* xn1_h  = (_Float16*)(ws + off); off += (size_t)RR * DM * 2;       // 16 MB
    _Float16* wqkv_t = (_Float16*)(ws + off); off += (size_t)C3 * DM * 2;       //  6 MB
    float*    bqkv   = (float*)   (ws + off); off += (size_t)C3 * 4;            // 12 KB
    _Float16* wo_t   = (_Float16*)(ws + off); off += (size_t)DM * DM * 2;       //  2 MB
    _Float16* w1_t   = (_Float16*)(ws + off); off += (size_t)DFF * DM * 2;      //  8 MB
    _Float16* w2_t   = (_Float16*)(ws + off); off += (size_t)DM * DFF * 2;      //  8 MB
    _Float16* qk_h   = (_Float16*)(ws + off); off += (size_t)RR * 2 * DM * 2;   // 32 MB
    _Float16* vt_h   = (_Float16*)(ws + off); off += (size_t)RR * DM * 2;       // 16 MB
    _Float16* o_h    = (_Float16*)(ws + off); off += (size_t)RR * DM * 2;       // 16 MB
    float*    x2     = (float*)   (ws + off); off += (size_t)RR * DM * 4;       // 32 MB
    _Float16* xn2_h  = (_Float16*)(ws + off); off += (size_t)RR * DM * 2;       // 16 MB
    _Float16* h_h    = (_Float16*)(ws + off); off += (size_t)RR * DFF * 2;      // 64 MB

    // 1. Pack weights to f16, transposed [N][K]
    pack_qkv_w_kernel<<<(C3 * DM) / 256, 256, 0, stream>>>(wq, wk, wv, wqkv_t);
    pack_qkv_b_kernel<<<(C3 + 255) / 256, 256, 0, stream>>>(bq, bk, bv, bqkv);
    cvt_t_kernel<<<(DM * DM) / 256, 256, 0, stream>>>(wo, wo_t, DM, DM);
    cvt_t_kernel<<<(DM * DFF) / 256, 256, 0, stream>>>(w1, w1_t, DM, DFF);   // -> [4096][1024]
    cvt_t_kernel<<<(DFF * DM) / 256, 256, 0, stream>>>(w2, w2_t, DFF, DM);   // -> [1024][4096]

    // 2. LN1
    ln_kernel<<<RR, 256, 0, stream>>>(x, ln1a, ln1b, xn1_h);

    // 3. QKV = xn1 @ [wq|wk|wv] + bias; Q|K -> qk_h, V -> vt_h (transposed)
    gemm_wmma_kernel<false, false, false, true>
        <<<dim3(C3 / 128, RR / 128), 256, 0, stream>>>(
        xn1_h, wqkv_t, bqkv, nullptr, (void*)qk_h, vt_h, RR, C3, DM);

    // 4. Attention (flash, online softmax)
    attn_kernel<<<BB * NH * (SS / 64), 128, 0, stream>>>(qk_h, vt_h, mask, o_h);

    // 5. x2 = x + o @ wo + bo
    gemm_wmma_kernel<false, true, true, false>
        <<<dim3(DM / 128, RR / 128), 256, 0, stream>>>(
        o_h, wo_t, bo, x, (void*)x2, nullptr, RR, DM, DM);

    // 6. LN2
    ln_kernel<<<RR, 256, 0, stream>>>(x2, ln2a, ln2b, xn2_h);

    // 7. h = relu(xn2 @ w1 + b1)
    gemm_wmma_kernel<true, false, false, false>
        <<<dim3(DFF / 128, RR / 128), 256, 0, stream>>>(
        xn2_h, w1_t, b1, nullptr, (void*)h_h, nullptr, RR, DFF, DM);

    // 8. out = x2 + h @ w2 + b2
    gemm_wmma_kernel<false, true, true, false>
        <<<dim3(DM / 128, RR / 128), 256, 0, stream>>>(
        h_h, w2_t, b2, x2, d_out, nullptr, RR, DM, DFF);
}
